// Attention_89421219102810
// MI455X (gfx1250) — matrix operand order repaired and checked by reading, not compiled, not run on hardware
//
#include <hip/hip_runtime.h>

// ---------------------------------------------------------------------------
// Types for CDNA5 WMMA (wave32, 16x16x32 bf16 -> f32)
// ---------------------------------------------------------------------------
typedef __attribute__((ext_vector_type(16))) __bf16 bf16x16;
typedef __attribute__((ext_vector_type(8)))  __bf16 bf16x8;
typedef __attribute__((ext_vector_type(8)))  float  f32x8;
typedef __attribute__((ext_vector_type(4)))  int    i32x4;

union Frag {
    bf16x16 v;
    bf16x8  h[2];
};

static __device__ __forceinline__ f32x8 wmma_bf16(const Frag& a, const Frag& b, f32x8 c) {
    // D = A(16x32 bf16) x B(32x16 bf16) + C(16x16 f32)
    return __builtin_amdgcn_wmma_f32_16x16x32_bf16(false, a.v, false, b.v,
                                                   (short)0, c, false, false);
}

// CDNA5 async global->LDS copy (ASYNCcnt-tracked). Probe round 2 revealed the
// builtin's first param is v4i addrspace(1)*; pass typed AS-qualified pointers.
#if defined(__has_builtin)
#if __has_builtin(__builtin_amdgcn_global_load_async_to_lds_b128)
#define HAVE_ASYNC_LDS 1
#endif
#endif

#if defined(HAVE_ASYNC_LDS)
typedef __attribute__((address_space(1))) i32x4* gv4i_ptr;
typedef __attribute__((address_space(3))) i32x4* lv4i_ptr;
static __device__ __forceinline__ void async_cp_b128(const __bf16* g, __bf16* l) {
    __builtin_amdgcn_global_load_async_to_lds_b128((gv4i_ptr)g, (lv4i_ptr)l, 0, 0);
}
#endif

// Problem constants (from the reference; SEQLENS is a compile-time constant)
#define S_TOT   2048
#define DIM     4096
#define NH      32
#define NKVH    8
#define HD      128
#define QROW    (NH * HD)     // 4096
#define KVROW   (NKVH * HD)   // 1024
#define ATTN_SCALE 0.08838834764831845f  // 128^-0.5

// ---------------------------------------------------------------------------
// 16-element staging: global (fp32 or bf16) -> registers -> LDS bf16
// ---------------------------------------------------------------------------
template <typename T> struct Raw16;
template <> struct Raw16<float>  { float4 f[4]; };
template <> struct Raw16<__bf16> { bf16x8 h[2]; };

template <typename T>
static __device__ __forceinline__ Raw16<T> ld16(const T* p);

template <>
__device__ __forceinline__ Raw16<float> ld16<float>(const float* p) {
    Raw16<float> r;
#pragma unroll
    for (int i = 0; i < 4; ++i) r.f[i] = *(const float4*)(p + 4 * i);
    return r;
}
template <>
__device__ __forceinline__ Raw16<__bf16> ld16<__bf16>(const __bf16* p) {
    Raw16<__bf16> r;
    r.h[0] = *(const bf16x8*)(p + 0);
    r.h[1] = *(const bf16x8*)(p + 8);
    return r;
}

static __device__ __forceinline__ void st16(const Raw16<float>& r, __bf16* dst) {
    union { bf16x8 v; __bf16 b[8]; } p0, p1;
    p0.b[0] = (__bf16)r.f[0].x; p0.b[1] = (__bf16)r.f[0].y;
    p0.b[2] = (__bf16)r.f[0].z; p0.b[3] = (__bf16)r.f[0].w;
    p0.b[4] = (__bf16)r.f[1].x; p0.b[5] = (__bf16)r.f[1].y;
    p0.b[6] = (__bf16)r.f[1].z; p0.b[7] = (__bf16)r.f[1].w;
    p1.b[0] = (__bf16)r.f[2].x; p1.b[1] = (__bf16)r.f[2].y;
    p1.b[2] = (__bf16)r.f[2].z; p1.b[3] = (__bf16)r.f[2].w;
    p1.b[4] = (__bf16)r.f[3].x; p1.b[5] = (__bf16)r.f[3].y;
    p1.b[6] = (__bf16)r.f[3].z; p1.b[7] = (__bf16)r.f[3].w;
    *(bf16x8*)(dst + 0) = p0.v;
    *(bf16x8*)(dst + 8) = p1.v;
}
static __device__ __forceinline__ void st16(const Raw16<__bf16>& r, __bf16* dst) {
    *(bf16x8*)(dst + 0) = r.h[0];
    *(bf16x8*)(dst + 8) = r.h[1];
}

// ---------------------------------------------------------------------------
// Double-buffered tiled GEMM: C[M,N](f32) = A[M,K] @ B[N,K]^T.
// A/B converted to bf16 while staging into LDS. 256 threads = 8 waves;
// 128x128 block tile, BK=32; each wave computes 4x2 16x16 WMMA tiles.
// LDS rows padded to 40 halves (80 B): 16B-aligned b128 frag loads, conflict-free.
// One barrier per K-step; next tile's global loads are issued before the WMMAs.
// ---------------------------------------------------------------------------
template <typename TA, typename TB>
__global__ __launch_bounds__(256) void gemm_bf16wmma(const TA* __restrict__ A,
                                                     const TB* __restrict__ B,
                                                     float* __restrict__ C,
                                                     int M, int N, int K) {
    __shared__ alignas(16) __bf16 As[2][128 * 40];
    __shared__ alignas(16) __bf16 Bs[2][128 * 40];

    const int tid  = threadIdx.x;
    const int lane = tid & 31;
    const int wave = tid >> 5;
    const int wr   = wave >> 2;        // 0..1 : 64-row slab
    const int wc   = wave & 3;         // 0..3 : 32-col slab
    const int hf   = lane >> 4;        // half-wave select
    const int ln   = lane & 15;

    const int m0 = blockIdx.y * 128;
    const int n0 = blockIdx.x * 128;

    // cooperative tile staging: thread t -> row t>>1, 16 cols at (t&1)*16
    const int lrow = tid >> 1;
    const int lcol = (tid & 1) * 16;
    const int lidx = lrow * 40 + lcol;
    const TA* ag = A + (size_t)(m0 + lrow) * K + lcol;
    const TB* bg = B + (size_t)(n0 + lrow) * K + lcol;

    f32x8 acc[4][2] = {};

    const int nk = K >> 5;
    // prologue: stage tile 0
    st16(ld16<TA>(ag), &As[0][lidx]);
    st16(ld16<TB>(bg), &Bs[0][lidx]);
    __syncthreads();

    for (int t = 0; t < nk; ++t) {
        const int  cur  = t & 1;
        const bool more = (t + 1) < nk;

        // issue next tile's global loads before the math
        Raw16<TA> na;
        Raw16<TB> nb;
        if (more) {
            na = ld16<TA>(ag + (size_t)(t + 1) * 32);
            nb = ld16<TB>(bg + (size_t)(t + 1) * 32);
            if (t + 4 < nk) {  // stream hint a few tiles ahead
                __builtin_prefetch(ag + (size_t)(t + 4) * 32, 0, 1);
                __builtin_prefetch(bg + (size_t)(t + 4) * 32, 0, 1);
            }
        }

        // A fragments: 16x32, lane<16 holds K {0..7,16..23}, lane>=16 K {8..15,24..31}
        Frag a[4], b[2];
        const int abase = hf * 8;
        const int bbase = hf * 8;
#pragma unroll
        for (int i = 0; i < 4; ++i) {
            const __bf16* p = &As[cur][(wr * 64 + i * 16 + ln) * 40 + abase];
            a[i].h[0] = *(const bf16x8*)p;
            a[i].h[1] = *(const bf16x8*)(p + 16);
        }
        // B fragments: 32x16, lane&15 = column, lane<16 K {0..7,16..23}, lane>=16 K {8..15,24..31}
#pragma unroll
        for (int j = 0; j < 2; ++j) {
            const __bf16* p = &Bs[cur][(wc * 32 + j * 16 + ln) * 40 + bbase];
            b[j].h[0] = *(const bf16x8*)p;
            b[j].h[1] = *(const bf16x8*)(p + 16);
        }
#pragma unroll
        for (int i = 0; i < 4; ++i)
#pragma unroll
            for (int j = 0; j < 2; ++j)
                acc[i][j] = wmma_bf16(a[i], b[j], acc[i][j]);

        if (more) {
            st16(na, &As[cur ^ 1][lidx]);
            st16(nb, &Bs[cur ^ 1][lidx]);
        }
        __syncthreads();
    }

    // C layout: VGPR r -> row r (lanes 0-15) / row r+8 (lanes 16-31), col = lane&15
#pragma unroll
    for (int i = 0; i < 4; ++i)
#pragma unroll
        for (int j = 0; j < 2; ++j) {
            const int row = m0 + wr * 64 + i * 16 + hf * 8;
            const int col = n0 + wc * 32 + j * 16 + ln;
#pragma unroll
            for (int r = 0; r < 8; ++r)
                C[(size_t)(row + r) * N + col] = acc[i][j][r];
        }
}

// ---------------------------------------------------------------------------
// RoPE (interleaved pairs) + fp32 -> bf16, scale folded in (Q gets 1/sqrt(D))
// t: (S, nheads, 128) f32 ; fc: (S, 64, 2) f32
// ---------------------------------------------------------------------------
__global__ void rope_bf16(const float* __restrict__ t, const float* __restrict__ fc,
                          __bf16* __restrict__ out, int nheads, int total, float scale) {
    int i = blockIdx.x * blockDim.x + threadIdx.x;
    if (i >= total) return;
    int d   = i & 63;
    int rem = i >> 6;
    int hd  = rem % nheads;
    int tok = rem / nheads;
    size_t base = ((size_t)tok * nheads + hd) * 128 + 2 * d;
    float a = t[base], b = t[base + 1];
    float c = fc[tok * 128 + 2 * d];
    float s = fc[tok * 128 + 2 * d + 1];
    out[base]     = (__bf16)((a * c - b * s) * scale);
    out[base + 1] = (__bf16)((a * s + b * c) * scale);
}

// fp32 -> bf16, 4 elements per thread (used for V)
__global__ void cvt_f32_bf16x4(const float* __restrict__ in, __bf16* __restrict__ out, int n4) {
    int i = blockIdx.x * blockDim.x + threadIdx.x;
    if (i >= n4) return;
    float4 f = ((const float4*)in)[i];
    union { __bf16 b[4]; ushort4 u; } o;
    o.b[0] = (__bf16)f.x; o.b[1] = (__bf16)f.y; o.b[2] = (__bf16)f.z; o.b[3] = (__bf16)f.w;
    ((ushort4*)out)[i] = o.u;
}

// ---------------------------------------------------------------------------
// Flash attention (varlen causal, GQA 4:1). Grid: (16 q-blocks of 128, 32 heads).
// Block = 256 threads = 8 waves; each wave owns a 16-row query tile, so every
// staged K/V tile is reused by 8 waves. K is staged with async global->LDS
// copies when available; V is staged transposed with packed b64 stores.
// Online softmax with 16-lane shuffle row reductions.
// ---------------------------------------------------------------------------
__global__ __launch_bounds__(256) void attn_kernel(const __bf16* __restrict__ qb,
                                                   const __bf16* __restrict__ kb_,
                                                   const __bf16* __restrict__ vb,
                                                   __bf16* __restrict__ ob) {
    __shared__ alignas(16) __bf16 Ks[32 * 136];    // [key token][d], stride 136 (272 B)
    __shared__ alignas(16) __bf16 Vt[128 * 40];    // [d][key token], stride 40 (80 B)
    __shared__ alignas(16) __bf16 Ps[8 * 16 * 40]; // per-wave P scratch (C->A relayout)

    const int tid  = threadIdx.x;
    const int lane = tid & 31;
    const int wave = tid >> 5;
    const int hf   = lane >> 4;
    const int ln   = lane & 15;

    const int q0  = blockIdx.x * 128;
    const int h   = blockIdx.y;
    const int kvh = h >> 2;   // REPEATS = 4

    // sequence boundaries: [0,512), [512,1280), [1280,1536), [1536,2048)
    int seqStart = 0;
    if (q0 >= 1536) seqStart = 1536;
    else if (q0 >= 1280) seqStart = 1280;
    else if (q0 >= 512)  seqStart = 512;

    const int qrow_base = q0 + wave * 16;
    const int myKEnd    = qrow_base + 16;

    // Q fragments (16x128 bf16, 4 K-chunks of 32), scale already folded in
    Frag aq[4];
    {
        const __bf16* qp = qb + (size_t)(qrow_base + ln) * QROW + h * HD;
        const int base = hf * 8;
#pragma unroll
        for (int c = 0; c < 4; ++c) {
            aq[c].h[0] = *(const bf16x8*)(qp + c * 32 + base);
            aq[c].h[1] = *(const bf16x8*)(qp + c * 32 + base + 16);
        }
    }

    f32x8 o[8] = {};
    float m_[8], l_[8];
#pragma unroll
    for (int r = 0; r < 8; ++r) { m_[r] = -3.0e38f; l_[r] = 0.0f; }

    const int kEnd = q0 + 128;
    for (int kb = seqStart; kb < kEnd; kb += 32) {
        // ---- K staging: thread t -> token t>>3, 16 d-vals at (t&7)*16
        {
            const int tok = tid >> 3;
            const int ch  = (tid & 7) * 16;
            const __bf16* kp = kb_ + (size_t)(kb + tok) * KVROW + kvh * HD + ch;
            __bf16* ks = &Ks[tok * 136 + ch];
#if defined(HAVE_ASYNC_LDS)
            async_cp_b128(kp, ks);
            async_cp_b128(kp + 8, ks + 8);
#else
            *(bf16x8*)(ks + 0) = *(const bf16x8*)(kp + 0);
            *(bf16x8*)(ks + 8) = *(const bf16x8*)(kp + 8);
#endif
        }
        // ---- V staging (transposed): thread t -> 4 tokens (t&7)*4, 4 d at (t>>3)*4;
        //      adjacent tokens are adjacent in Vt rows -> packed b64 stores.
        {
            const int tg = (tid & 7) * 4;
            const int dr = (tid >> 3) * 4;
            const __bf16* vpb = vb + kvh * HD + dr;
            union V4 { uint2 u; __bf16 b[4]; } vv[4], pk;
#pragma unroll
            for (int q = 0; q < 4; ++q)
                vv[q].u = *(const uint2*)(vpb + (size_t)(kb + tg + q) * KVROW);
#pragma unroll
            for (int e = 0; e < 4; ++e) {
                pk.b[0] = vv[0].b[e]; pk.b[1] = vv[1].b[e];
                pk.b[2] = vv[2].b[e]; pk.b[3] = vv[3].b[e];
                *(uint2*)&Vt[(dr + e) * 40 + tg] = pk.u;
            }
        }
#if defined(HAVE_ASYNC_LDS)
        asm volatile("s_wait_asynccnt 0x0" ::: "memory");
#endif
        __syncthreads();

        if (kb < myKEnd) {  // wave-uniform causal skip
            // ---- scores: S = Q @ K^T, two 16-key tiles, K-dim = 128 in 4 chunks
            f32x8 st[2] = {};
#pragma unroll
            for (int j = 0; j < 2; ++j) {
                const __bf16* krow = &Ks[(j * 16 + ln) * 136 + hf * 8];
#pragma unroll
                for (int c = 0; c < 4; ++c) {
                    Frag bk;
                    bk.h[0] = *(const bf16x8*)(krow + c * 32);
                    bk.h[1] = *(const bf16x8*)(krow + c * 32 + 16);
                    st[j] = wmma_bf16(aq[c], bk, st[j]);
                }
            }
            // ---- mask + online softmax (row stats via 16-lane shuffles)
            const int qrow = qrow_base + hf * 8;
#pragma unroll
            for (int r = 0; r < 8; ++r) {
                const int qr = qrow + r;
                float s0 = (kb + ln      <= qr) ? st[0][r] : -3.0e38f;
                float s1 = (kb + 16 + ln <= qr) ? st[1][r] : -3.0e38f;
                float mx = fmaxf(s0, s1);
#pragma unroll
                for (int off = 1; off < 16; off <<= 1)
                    mx = fmaxf(mx, __shfl_xor(mx, off, 32));
                const float mnew = fmaxf(m_[r], mx);
                const float sc   = __expf(m_[r] - mnew);
                const float p0   = __expf(s0 - mnew);
                const float p1   = __expf(s1 - mnew);
                float sum = p0 + p1;
#pragma unroll
                for (int off = 1; off < 16; off <<= 1)
                    sum += __shfl_xor(sum, off, 32);
                l_[r] = l_[r] * sc + sum;
                m_[r] = mnew;
#pragma unroll
                for (int dt = 0; dt < 8; ++dt) o[dt][r] *= sc;
                // write P in C-layout to per-wave LDS scratch
                __bf16* pp = &Ps[wave * 640 + (hf * 8 + r) * 40];
                pp[ln]      = (__bf16)p0;
                pp[16 + ln] = (__bf16)p1;
            }
            // same-wave LDS RAW: ensure ds stores complete before fragment reads
            asm volatile("s_wait_dscnt 0x0" ::: "memory");

            // ---- P as A-fragment (16 rows x 32 keys)
            Frag ap;
            {
                const __bf16* pr = &Ps[wave * 640 + ln * 40 + hf * 8];
                ap.h[0] = *(const bf16x8*)pr;
                ap.h[1] = *(const bf16x8*)(pr + 16);
            }
            // ---- O += P @ V  (V^T staged: B-frag = V[kb+k][dcol], k = 8*hf+{0..7,16..23})
#pragma unroll
            for (int dt = 0; dt < 8; ++dt) {
                const __bf16* vrow = &Vt[(dt * 16 + ln) * 40 + hf * 8];
                Frag bv;
                bv.h[0] = *(const bf16x8*)vrow;
                bv.h[1] = *(const bf16x8*)(vrow + 16);
                o[dt] = wmma_bf16(ap, bv, o[dt]);
            }
        }
        __syncthreads();
    }

    // ---- normalize and emit bf16 attention output (token, h*128 + d)
#pragma unroll
    for (int r = 0; r < 8; ++r) {
        const float inv = 1.0f / l_[r];
        const int row = qrow_base + hf * 8 + r;
        __bf16* op = ob + (size_t)row * QROW + h * HD;
#pragma unroll
        for (int dt = 0; dt < 8; ++dt)
            op[dt * 16 + ln] = (__bf16)(o[dt][r] * inv);
    }
}

// ---------------------------------------------------------------------------
// Orchestration
// ---------------------------------------------------------------------------
extern "C" void kernel_launch(void* const* d_in, const int* in_sizes, int n_in,
                              void* d_out, int out_size, void* d_ws, size_t ws_size,
                              hipStream_t stream) {
    (void)in_sizes; (void)n_in; (void)out_size; (void)ws_size;
    const float* x  = (const float*)d_in[0];
    const float* fc = (const float*)d_in[1];
    // d_in[2] = seqlens (compile-time constant in the reference; hardcoded)
    const float* wq = (const float*)d_in[3];
    const float* wk = (const float*)d_in[4];
    const float* wv = (const float*)d_in[5];
    const float* wo = (const float*)d_in[6];
    float* out = (float*)d_out;

    char* w = (char*)d_ws;
    auto alloc = [&](size_t bytes) {
        char* p = w;
        w += (bytes + 255) & ~(size_t)255;
        return p;
    };
    float*  qf    = (float*) alloc((size_t)S_TOT * QROW  * 4);  // Q proj (f32)
    float*  kf    = (float*) alloc((size_t)S_TOT * KVROW * 4);  // K proj (f32)
    float*  vf    = (float*) alloc((size_t)S_TOT * KVROW * 4);  // V proj (f32)
    __bf16* qbuf  = (__bf16*)alloc((size_t)S_TOT * QROW  * 2);  // Q post-RoPE*scale
    __bf16* kbuf  = (__bf16*)alloc((size_t)S_TOT * KVROW * 2);  // K post-RoPE
    __bf16* vbuf  = (__bf16*)alloc((size_t)S_TOT * KVROW * 2);  // V bf16
    __bf16* attnb = (__bf16*)alloc((size_t)S_TOT * QROW  * 2);  // attention output bf16

    // QKV projections: C = x @ W^T ; W is [N, K=DIM] row-major, converted in-kernel
    gemm_bf16wmma<float, float><<<dim3(QROW / 128,  S_TOT / 128), 256, 0, stream>>>(
        x, wq, qf, S_TOT, QROW, DIM);
    gemm_bf16wmma<float, float><<<dim3(KVROW / 128, S_TOT / 128), 256, 0, stream>>>(
        x, wk, kf, S_TOT, KVROW, DIM);
    gemm_bf16wmma<float, float><<<dim3(KVROW / 128, S_TOT / 128), 256, 0, stream>>>(
        x, wv, vf, S_TOT, KVROW, DIM);

    // RoPE + bf16 (scale folded into Q); V straight convert
    {
        int tq = S_TOT * NH * 64;
        rope_bf16<<<tq / 256, 256, 0, stream>>>(qf, fc, qbuf, NH, tq, ATTN_SCALE);
        int tk = S_TOT * NKVH * 64;
        rope_bf16<<<tk / 256, 256, 0, stream>>>(kf, fc, kbuf, NKVH, tk, 1.0f);
        int nv4 = (S_TOT * KVROW) / 4;
        cvt_f32_bf16x4<<<nv4 / 256, 256, 0, stream>>>(vf, vbuf, nv4);
    }

    // Flash attention: 16 q-blocks x 32 heads
    attn_kernel<<<dim3(S_TOT / 128, NH), 256, 0, stream>>>(qbuf, kbuf, vbuf, attnb);

    // Output projection: out = attn @ wo^T
    gemm_bf16wmma<__bf16, float><<<dim3(DIM / 128, S_TOT / 128), 256, 0, stream>>>(
        attnb, wo, out, S_TOT, DIM, QROW);
}
